// GCN_E_85358180041299
// MI455X (gfx1250) — matrix operand order repaired and checked by reading, not compiled, not run on hardware
//
#include <hip/hip_runtime.h>
#include <hip/hip_bf16.h>

typedef __attribute__((ext_vector_type(8)))  __bf16 bf16x8;
typedef __attribute__((ext_vector_type(16))) __bf16 bf16x16;
typedef __attribute__((ext_vector_type(8)))  float  f32x8;

static __device__ __forceinline__ unsigned short f32_to_bf16_rne(float f) {
  union { float f; unsigned int u; } c; c.f = f;
  unsigned int u = c.u;
  u += 0x7FFFu + ((u >> 16) & 1u);
  return (unsigned short)(u >> 16);
}

// ---------------- conversion / prep kernels ----------------

__global__ void cvt_f32_to_bf16(const float* __restrict__ in,
                                unsigned short* __restrict__ out, long n) {
  long i = (long)blockIdx.x * blockDim.x + threadIdx.x;
  if (i < n) out[i] = f32_to_bf16_rne(in[i]);
}

// out[f*K + k] = (f < F) ? W[(k0+k)*F + f] : 0, for f in [0, Fpad)
__global__ void wprep_transpose_cvt(const float* __restrict__ W,
                                    unsigned short* __restrict__ out,
                                    int K, int F, int k0, int Fpad) {
  long i = (long)blockIdx.x * blockDim.x + threadIdx.x;
  long n = (long)Fpad * K;
  if (i >= n) return;
  int f = (int)(i / K);
  int k = (int)(i % K);
  float v = (f < F) ? W[(long)(k0 + k) * F + f] : 0.0f;
  out[i] = f32_to_bf16_rne(v);
}

// [B, R, C] -> [B, C, R] bf16, LDS-tiled (R, C multiples of 32)
__global__ void transpose_bf16(const unsigned short* __restrict__ in,
                               unsigned short* __restrict__ out, int R, int C) {
  __shared__ unsigned short t[32][33];
  const int b  = blockIdx.z;
  const int c0 = blockIdx.x * 32, r0 = blockIdx.y * 32;
  const unsigned short* ib = in  + (long)b * R * C;
  unsigned short*       ob = out + (long)b * R * C;
#pragma unroll
  for (int i = 0; i < 4; ++i)
    t[threadIdx.y + 8 * i][threadIdx.x] =
        ib[(long)(r0 + threadIdx.y + 8 * i) * C + c0 + threadIdx.x];
  __syncthreads();
#pragma unroll
  for (int i = 0; i < 4; ++i)
    ob[(long)(c0 + threadIdx.y + 8 * i) * R + r0 + threadIdx.x] =
        t[threadIdx.x][threadIdx.y + 8 * i];
}

// ---------------- bf16 WMMA GEMM ----------------
// C[b, m, n] = sum_k A[b,m,k]*Bt[n,k]  (+ A2[b,m,k]*Bt2[n,k] if DUAL)
// A,A2: [batch, M, K] bf16 row-major. Bt,Bt2: [F, K] bf16 ("transposed B").
// Epilogue: + bias[n]; relu OR channelwise PReLU(alpha[n]); write bf16 or f32.
// Grid: (F/128, M/256, batch). Block: 256 = 8 waves; wave -> 64x64 tile.
// Software-pipelined: chunk = one 32-deep k-slice of one source; double-buffered
// operand sets so chunk c+1 loads overlap chunk c's 16 WMMAs.
template <bool DUAL>
__global__ __launch_bounds__(256, 1)
void gemm_wmma_bf16(const __bf16* __restrict__ A,  const __bf16* __restrict__ A2,
                    const __bf16* __restrict__ Bt, const __bf16* __restrict__ Bt2,
                    const float* __restrict__ bias, const float* __restrict__ alpha,
                    unsigned short* __restrict__ outB, float* __restrict__ outF,
                    int M, int K,
                    long sA, long sBt, long sOut,
                    int ldOut, int Fstore, int doRelu) {
  const int lane  = threadIdx.x & 31;
  const int wave  = threadIdx.x >> 5;
  const int wm    = wave & 3, wn = wave >> 2;        // 4 waves along M, 2 along N
  const int mBase = blockIdx.y * 256 + wm * 64;
  const int nBase = blockIdx.x * 128 + wn * 64;
  const int b     = blockIdx.z;
  const int row   = lane & 15;
  const int hi    = lane >> 4;

  // Uniform (SGPR) batch bases; per-lane 32-bit element offsets.
  const __bf16* Ab   = A + (long)b * sA;
  const __bf16* A2b  = DUAL ? A2 + (long)b * sA : nullptr;
  const __bf16* Btb  = Bt + (long)b * sBt;
  const __bf16* Bt2b = DUAL ? Bt2 + (long)b * sBt : nullptr;

  unsigned aOff[4], bOff[4];
#pragma unroll
  for (int t = 0; t < 4; ++t) {
    aOff[t] = (unsigned)(mBase + t * 16 + row) * (unsigned)K + (unsigned)(hi << 4);
    bOff[t] = (unsigned)(nBase + t * 16 + row) * (unsigned)K + (unsigned)(hi << 4);
  }

  f32x8 acc[4][4] = {};

  // chunk c: source = (DUAL ? c&1 : 0), k-slice = (DUAL ? c>>1 : c) * 32
  const int NC = (DUAL ? 2 : 1) * (K >> 5);

  auto srcA = [&](int c) -> const __bf16* { return (DUAL && (c & 1)) ? A2b  : Ab;  };
  auto srcB = [&](int c) -> const __bf16* { return (DUAL && (c & 1)) ? Bt2b : Btb; };
  auto kOf  = [&](int c) -> unsigned { return (unsigned)((DUAL ? (c >> 1) : c) << 5); };

  auto loadTiles = [&](bf16x16* av, bf16x16* bv, const __bf16* As, const __bf16* Bs,
                       unsigned kk) {
#pragma unroll
    for (int t = 0; t < 4; ++t) {
      const __bf16* p = As + aOff[t] + kk;
      bf16x8 lo = *(const bf16x8*)p;
      bf16x8 hh = *(const bf16x8*)(p + 8);
      av[t] = __builtin_shufflevector(lo, hh, 0,1,2,3,4,5,6,7,8,9,10,11,12,13,14,15);
    }
#pragma unroll
    for (int t = 0; t < 4; ++t)
      bv[t] = *(const bf16x16*)(Bs + bOff[t] + kk);
  };
  auto mac = [&](const bf16x16* av, const bf16x16* bv) {
#pragma unroll
    for (int i = 0; i < 4; ++i)
#pragma unroll
      for (int j = 0; j < 4; ++j)
        acc[i][j] = __builtin_amdgcn_wmma_f32_16x16x32_bf16(
            false, av[i], false, bv[j], (short)0, acc[i][j], false, false);
  };

  bf16x16 av0[4], bv0[4], av1[4], bv1[4];
  loadTiles(av0, bv0, srcA(0), srcB(0), kOf(0));
  for (int c = 0; c < NC; c += 2) {   // NC is always even for our shapes
    loadTiles(av1, bv1, srcA(c + 1), srcB(c + 1), kOf(c + 1));
    mac(av0, bv0);
    if (c + 2 < NC)
      loadTiles(av0, bv0, srcA(c + 2), srcB(c + 2), kOf(c + 2));
    mac(av1, bv1);
  }

  // Epilogue. C/D layout: vgpr r of a 16x16 tile = row r (+8 for lanes 16-31), col = lane&15.
  const long outBatch = (long)b * sOut;
#pragma unroll
  for (int i = 0; i < 4; ++i) {
#pragma unroll
    for (int j = 0; j < 4; ++j) {
      const int n  = nBase + j * 16 + (lane & 15);
      const int mb = mBase + i * 16 + (hi << 3);
      const bool nOK = (n < Fstore);
      const float bvs = (bias  && nOK) ? bias[n]  : 0.0f;
      const float avs = (alpha && nOK) ? alpha[n] : 0.0f;
#pragma unroll
      for (int r = 0; r < 8; ++r) {
        const int m = mb + r;
        float v = acc[i][j][r] + bvs;
        if (doRelu)      v = v > 0.0f ? v : 0.0f;
        else if (alpha)  v = v > 0.0f ? v : avs * v;
        if (m < M && nOK) {
          const long o = outBatch + (long)m * ldOut + n;
          if (outB) outB[o] = f32_to_bf16_rne(v);
          else      outF[o] = v;
        }
      }
    }
  }
}

// ---------------- host orchestration ----------------

extern "C" void kernel_launch(void* const* d_in, const int* in_sizes, int n_in,
                              void* d_out, int out_size, void* d_ws, size_t ws_size,
                              hipStream_t stream) {
  (void)in_sizes; (void)n_in; (void)out_size; (void)ws_size;
  const int B = 32, N = 1024, D = 256;

  const float* x     = (const float*)d_in[0];
  const float* adj   = (const float*)d_in[1];
  const float* W1    = (const float*)d_in[2];
  const float* b1    = (const float*)d_in[3];
  const float* W2    = (const float*)d_in[4];
  const float* b2    = (const float*)d_in[5];
  const float* W3    = (const float*)d_in[6];
  const float* b3    = (const float*)d_in[7];
  const float* W4    = (const float*)d_in[8];
  const float* b4    = (const float*)d_in[9];
  const float* cW1   = (const float*)d_in[10];
  const float* cb1   = (const float*)d_in[11];
  const float* alpha = (const float*)d_in[12];
  const float* cW2   = (const float*)d_in[13];
  const float* cb2   = (const float*)d_in[14];

  char* wsp = (char*)d_ws;
  auto carve = [&](size_t bytes) -> char* {
    char* p = wsp;
    wsp += (bytes + 255) & ~(size_t)255;
    return p;
  };
  unsigned short* adjB = (unsigned short*)carve((size_t)B * N * N * 2);   // 64 MB
  unsigned short* P0   = (unsigned short*)carve((size_t)B * N * 512 * 2); // 32 MB
  unsigned short* P1   = (unsigned short*)carve((size_t)B * N * 512 * 2); // 32 MB
  unsigned short* AG   = (unsigned short*)carve((size_t)B * N * 512 * 2); // 32 MB
  unsigned short* W1tT = (unsigned short*)carve((size_t)512 * 256 * 2);
  unsigned short* W1tB = (unsigned short*)carve((size_t)512 * 256 * 2);
  unsigned short* W2tT = (unsigned short*)carve((size_t)512 * 512 * 2);
  unsigned short* W2tB = (unsigned short*)carve((size_t)512 * 512 * 2);
  unsigned short* W3tT = (unsigned short*)carve((size_t)256 * 512 * 2);
  unsigned short* W3tB = (unsigned short*)carve((size_t)256 * 512 * 2);
  unsigned short* W4tT = (unsigned short*)carve((size_t)256 * 256 * 2);
  unsigned short* W4tB = (unsigned short*)carve((size_t)256 * 256 * 2);
  unsigned short* C1t  = (unsigned short*)carve((size_t)256 * 256 * 2);
  unsigned short* C2t  = (unsigned short*)carve((size_t)128 * 256 * 2);

  auto cdiv = [](long a, long b) { return (a + b - 1) / b; };

  // Convert adj and x to bf16.
  {
    long n = (long)B * N * N;
    cvt_f32_to_bf16<<<dim3((unsigned)cdiv(n, 256)), 256, 0, stream>>>(adj, adjB, n);
    long nx = (long)B * N * D;
    cvt_f32_to_bf16<<<dim3((unsigned)cdiv(nx, 256)), 256, 0, stream>>>(x, P0, nx);
  }
  // Weight prep: W[(k0+k)*F + f] -> Wt[f*K + k] (bf16), zero-padded to Fpad rows.
  auto wprep = [&](const float* W, unsigned short* out, int K, int F, int k0, int Fpad) {
    long n = (long)Fpad * K;
    wprep_transpose_cvt<<<dim3((unsigned)cdiv(n, 256)), 256, 0, stream>>>(W, out, K, F, k0, Fpad);
  };
  wprep(W1, W1tT, 256, 512, 0, 512);  wprep(W1, W1tB, 256, 512, 256, 512);
  wprep(W2, W2tT, 512, 512, 0, 512);  wprep(W2, W2tB, 512, 512, 512, 512);
  wprep(W3, W3tT, 512, 256, 0, 256);  wprep(W3, W3tB, 512, 256, 512, 256);
  wprep(W4, W4tT, 256, 256, 0, 256);  wprep(W4, W4tB, 256, 256, 256, 256);
  wprep(cW1, C1t, 256, 256, 0, 256);
  wprep(cW2, C2t, 256, 2,   0, 128);

  struct Layer { int Kd, Fd; unsigned short *wt, *wb; const float* bias; };
  Layer L[4] = {
    {256, 512, W1tT, W1tB, b1},
    {512, 512, W2tT, W2tB, b2},
    {512, 256, W3tT, W3tB, b3},
    {256, 256, W4tT, W4tB, b4},
  };

  unsigned short* hcur = P0;
  unsigned short* hsc  = P1;
  for (int l = 0; l < 4; ++l) {
    const int Kd = L[l].Kd, Fd = L[l].Fd;
    // hT = transpose(h): [B, N, Kd] -> [B, Kd, N]
    transpose_bf16<<<dim3(Kd / 32, N / 32, B), dim3(32, 8), 0, stream>>>(hcur, hsc, N, Kd);
    // agg = adj @ h  (A = adj [N,N], Bt = hT [Kd, N], K = N)
    gemm_wmma_bf16<false><<<dim3(Kd / 128, N / 256, B), 256, 0, stream>>>(
        (const __bf16*)adjB, nullptr, (const __bf16*)hsc, nullptr,
        nullptr, nullptr, AG, nullptr,
        N, N,
        (long)N * N, (long)Kd * N, (long)N * Kd,
        Kd, Kd, 0);
    // h_next = relu(h @ Wtop + agg @ Wbot + bias) -> hsc (hT dead now)
    gemm_wmma_bf16<true><<<dim3(Fd / 128, N / 256, B), 256, 0, stream>>>(
        (const __bf16*)hcur, (const __bf16*)AG,
        (const __bf16*)L[l].wt, (const __bf16*)L[l].wb,
        L[l].bias, nullptr, hsc, nullptr,
        N, Kd,
        (long)N * Kd, 0L, (long)N * Fd,
        Fd, Fd, 1);
    unsigned short* t = hcur; hcur = hsc; hsc = t;
  }

  // Classifier: t = PReLU(h @ cW1 + cb1)
  gemm_wmma_bf16<false><<<dim3(2, N / 256, B), 256, 0, stream>>>(
      (const __bf16*)hcur, nullptr, (const __bf16*)C1t, nullptr,
      cb1, alpha, AG, nullptr,
      N, 256,
      (long)N * 256, 0L, (long)N * 256,
      256, 256, 0);
  // pred = t @ cW2 + cb2 -> f32 d_out [B, N, 2] (F padded to 128, stores guarded)
  gemm_wmma_bf16<false><<<dim3(1, N / 256, B), 256, 0, stream>>>(
      (const __bf16*)AG, nullptr, (const __bf16*)C2t, nullptr,
      cb2, nullptr, nullptr, (float*)d_out,
      N, 256,
      (long)N * 256, 0L, (long)N * 2,
      2, 2, 0);
}
